// LSTMDecoder_recurrent_35794257444983
// MI455X (gfx1250) — compile-verified, matrix operand order checked by reading, not run on hardware
//
#include <hip/hip_runtime.h>
#include <hip/hip_bf16.h>

#define SEQ_LEN 181
#define NB      4096
#define H       128
#define OUTF    12

typedef __attribute__((ext_vector_type(16))) __bf16 v16bf;
typedef __attribute__((ext_vector_type(8)))  float  v8f;
typedef __attribute__((ext_vector_type(4)))  unsigned int v4u;
typedef __attribute__((ext_vector_type(8)))  int v8i_;
typedef __attribute__((ext_vector_type(4)))  int v4i_;

// ---------------- workspace layout (bytes) ----------------
// [0)       layer0 packed weights: Wih0 (131072) + Whh0 (131072)
// [262144)  layer1 packed weights: Wih1 + Whh1
// [524288)  WoutT packed: 8 frags * 1024B
// [532480)  bsum0 (512 f32) | [534528) bsum1 (512 f32) | [536576) bout16 (16 f32)
static constexpr int WS_W0   = 0;
static constexpr int WS_W1   = 262144;
static constexpr int WS_WOUT = 524288;
static constexpr int WS_BS0  = 532480;

// ---------------- helpers ----------------
__device__ __forceinline__ unsigned short f2bf(float f) {
  unsigned u = __float_as_uint(f);
  u += 0x7fffu + ((u >> 16) & 1u);            // round-to-nearest-even
  return (unsigned short)(u >> 16);
}
__device__ __forceinline__ float sigm(float x) { return 1.f / (1.f + __expf(-x)); }
__device__ __forceinline__ float tanh_f(float x) {
  float a = fabsf(x);
  float t = __expf(-2.f * a);
  float r = (1.f - t) / (1.f + t);
  return (x < 0.f) ? -r : r;
}
__device__ __forceinline__ v8f splat8(float b) { v8f r = {b,b,b,b,b,b,b,b}; return r; }

// 1-D TDM DMA: global -> LDS, nbytes must be a multiple of 8 and <= 512KB.
// D# per CDNA5 ISA ch.8: group0 = {count, lds_addr, global_addr, type=2},
// group1 = {data_size=8B, tensor_dim0 = tile_dim0 = n8, dim1 = 1, stride0 = n8}.
__device__ __forceinline__ void tdm_load_1d(unsigned lds_addr, const void* gaddr, unsigned nbytes) {
  unsigned long long ga = (unsigned long long)(uintptr_t)gaddr;
  unsigned n8 = nbytes >> 3;
  v4u g0;
  g0[0] = 1u;                                           // count=1, user descriptor
  g0[1] = lds_addr;                                     // LDS byte address
  g0[2] = (unsigned)(ga & 0xffffffffu);                 // global_addr[31:0]
  g0[3] = (unsigned)((ga >> 32) & 0x01ffffffu) | (2u << 30);  // addr[56:32] | type=2
  v8i_ g1;
  g1[0] = (int)(3u << 16);                              // wg_mask=0, data_size=3 (8B)
  g1[1] = (int)((n8 & 0xffffu) << 16);                  // tensor_dim0[15:0]  @ [63:48]
  g1[2] = (int)((n8 >> 16) & 0xffffu) | (int)(1u << 16);// tensor_dim0[31:16] | tensor_dim1=1
  g1[3] = (int)((n8 & 0xffffu) << 16);                  // tile_dim0 @ [127:112]
  g1[4] = 1;                                            // tile_dim1 = 1
  g1[5] = (int)n8;                                      // tensor_dim0_stride[31:0]
  g1[6] = 0;
  g1[7] = 0;
  v4i_ z4 = {0, 0, 0, 0};
  v8i_ z8 = {0, 0, 0, 0, 0, 0, 0, 0};
  // 6-arg toolchain signature: (g0, g1, g2, g3, g4, cpol)
  __builtin_amdgcn_tensor_load_to_lds(g0, g1, z4, z4, z8, 0);
}

// A-fragment (16x32 bf16, M x K) from a row-major 16x128 bf16 LDS buffer.
__device__ __forceinline__ v16bf loadA_lds(const unsigned short* buf, int kb, int lane) {
  int row = lane & 15;
  int sub = (lane < 16) ? 0 : 8;
  const unsigned short* p = buf + row * H + kb * 32 + sub;
  v16bf a;
  ((uint4*)&a)[0] = *(const uint4*)(p);
  ((uint4*)&a)[1] = *(const uint4*)(p + 16);
  return a;
}
// B-fragment from pre-packed fragment array: 32B contiguous per (frag,lane)
__device__ __forceinline__ v16bf loadB_pk(const unsigned short* wm, int frag, int lane) {
  const unsigned short* p = wm + (frag * 32 + lane) * 16;
  v16bf b;
  ((uint4*)&b)[0] = *(const uint4*)(p);
  ((uint4*)&b)[1] = *(const uint4*)(p + 8);
  return b;
}

// ---------------- setup kernels ----------------
__global__ void pack_gates(const float* __restrict__ Wih0, const float* __restrict__ Whh0,
                           const float* __restrict__ Wih1, const float* __restrict__ Whh1,
                           unsigned short* __restrict__ dst) {
  int idx  = blockIdx.x * 256 + threadIdx.x;   // < 262144
  int mat  = idx >> 16;
  int r    = idx & 65535;
  int frag = r >> 9;
  int lane = (r >> 4) & 31;
  int e    = r & 15;
  int tile = frag >> 2, kb = frag & 3;
  int n    = tile * 16 + (lane & 15);
  int sub  = (lane < 16) ? 0 : 8;
  int k    = kb * 32 + ((e < 8) ? (sub + e) : (16 + sub + (e - 8)));
  const float* W = (mat == 0) ? Wih0 : (mat == 1) ? Whh0 : (mat == 2) ? Wih1 : Whh1;
  dst[idx] = f2bf(W[n * H + k]);
}

__global__ void pack_wout(const float* __restrict__ Wout, unsigned short* __restrict__ dst) {
  int idx  = blockIdx.x * 256 + threadIdx.x;   // < 4096
  int frag = idx >> 9;
  int lane = (idx >> 4) & 31;
  int e    = idx & 15;
  int n    = lane & 15;
  int sub  = (lane < 16) ? 0 : 8;
  int k    = frag * 32 + ((e < 8) ? (sub + e) : (16 + sub + (e - 8)));
  float v  = (n < OUTF) ? Wout[n * (2 * H) + k] : 0.f;
  dst[idx] = f2bf(v);
}

__global__ void pack_bias(const float* __restrict__ bih0, const float* __restrict__ bhh0,
                          const float* __restrict__ bih1, const float* __restrict__ bhh1,
                          const float* __restrict__ bout, float* __restrict__ bsum) {
  int i = blockIdx.x * 256 + threadIdx.x;
  if (i < 512)       bsum[i] = bih0[i] + bhh0[i];
  else if (i < 1024) bsum[i] = bih1[i - 512] + bhh1[i - 512];
  else if (i < 1040) { int j = i - 1024; bsum[i] = (j < OUTF) ? bout[j] : 0.f; }
}

// ---------------- LSTM cell ----------------
template <bool WLDS>
__device__ __forceinline__ void lstm_cell(
    int lane, int wv,
    const unsigned short* xBuf, const unsigned short* hBuf,   // LDS 16x128 bf16
    const unsigned short* wIh, const unsigned short* wHh,     // packed LDS weights (WLDS)
    const v16bf (*wIhR)[4], const v16bf (*wHhR)[4],           // register weights (!WLDS)
    const float* bias,                                        // LDS bsum (512)
    v8f& cacc,
    unsigned short* out0, unsigned short* out1) {
  v16bf ax[4], ah[4];
#pragma unroll
  for (int kb = 0; kb < 4; ++kb) {
    ax[kb] = loadA_lds(xBuf, kb, lane);
    ah[kb] = loadA_lds(hBuf, kb, lane);
  }
  v8f acc[4];
#pragma unroll
  for (int t = 0; t < 4; ++t) {                         // t: 0=i 1=f 2=g 3=o
    int tile = wv + 8 * t;
    acc[t] = splat8(bias[t * 128 + wv * 16 + (lane & 15)]);
#pragma unroll
    for (int kb = 0; kb < 4; ++kb) {
      v16bf bi, bh;
      if constexpr (WLDS) {
        bi = loadB_pk(wIh, tile * 4 + kb, lane);
        bh = loadB_pk(wHh, tile * 4 + kb, lane);
      } else {
        bi = wIhR[t][kb];
        bh = wHhR[t][kb];
      }
      acc[t] = __builtin_amdgcn_wmma_f32_16x16x32_bf16(false, ax[kb], false, bi,
                                                       (short)0, acc[t], false, false);
      acc[t] = __builtin_amdgcn_wmma_f32_16x16x32_bf16(false, ah[kb], false, bh,
                                                       (short)0, acc[t], false, false);
    }
  }
  int col   = wv * 16 + (lane & 15);
  int rbase = (lane < 16) ? 0 : 8;
#pragma unroll
  for (int v = 0; v < 8; ++v) {
    float ig = sigm(acc[0][v]);
    float fg = sigm(acc[1][v]);
    float gg = tanh_f(acc[2][v]);
    float og = sigm(acc[3][v]);
    float c  = fg * cacc[v] + ig * gg;
    cacc[v]  = c;
    float h  = og * tanh_f(c);
    unsigned short hb = f2bf(h);
    out0[(rbase + v) * H + col] = hb;
    if (out1) out1[(rbase + v) * H + col] = hb;
  }
}

// ---------------- main persistent kernel ----------------
// one block = 16 batch rows for all 181 steps; 8 waves; wave w owns gate cols [16w,16w+16)
__global__ __launch_bounds__(256, 1)
void lstm_main(const float* __restrict__ enc_h, const float* __restrict__ enc_c,
               const unsigned char* __restrict__ ws, float* __restrict__ out) {
  __shared__ __align__(16) unsigned char smem[294976];
  unsigned short* sW0   = (unsigned short*)(smem + 0);        // Wih0 | Whh0 (256KB)
  unsigned short* sWout = (unsigned short*)(smem + 262144);   // 8KB
  float*          sBS0  = (float*)(smem + 270336);            // 2KB
  float*          sBS1  = (float*)(smem + 272384);            // 2KB
  float*          sBout = (float*)(smem + 274432);            // 64B
  unsigned short* bufA  = (unsigned short*)(smem + 274496);   // dec_in0 / L1-t0 out
  unsigned short* bufB  = (unsigned short*)(smem + 278592);   // dec_in1 / L1-t1 out
  unsigned short* bufM0 = (unsigned short*)(smem + 282688);   // L0-t0 out
  unsigned short* bufM1 = (unsigned short*)(smem + 286784);   // L0-t1 out == h0 state
  unsigned short* bufH1 = (unsigned short*)(smem + 290880);   // h1 state

  const int tid  = threadIdx.x;
  const int lane = tid & 31;
  const int wv   = tid >> 5;
  const int b0   = blockIdx.x * 16;

  // --- async TDM staging of L0 weights + WoutT + fused biases into LDS ---
  // (overlaps with register-weight loads and buffer init below)
  if (wv == 0) {
    unsigned lb = (unsigned)(uintptr_t)smem;   // low 32 bits of generic ptr == LDS offset
    tdm_load_1d(lb + 0,      ws + WS_W0,   262144);  // Wih0|Whh0
    tdm_load_1d(lb + 262144, ws + WS_WOUT, 8192);    // WoutT
    tdm_load_1d(lb + 270336, ws + WS_BS0,  4160);    // bsum0|bsum1|bout16
  }

  // init activation buffers: dec_in = 0, h0 = enc_h[0], h1 = enc_h[1]
  for (int i = tid; i < 16 * H; i += 256) {
    int r = i >> 7, c = i & 127;
    bufA[i]  = 0;
    bufB[i]  = 0;
    bufM1[i] = f2bf(enc_h[0 * NB * H + (b0 + r) * H + c]);
    bufH1[i] = f2bf(enc_h[1 * NB * H + (b0 + r) * H + c]);
  }

  // c state in registers (WMMA C/D layout)
  v8f c0acc, c1acc;
  {
    int col = wv * 16 + (lane & 15);
    int rb  = (lane < 16) ? 0 : 8;
#pragma unroll
    for (int v = 0; v < 8; ++v) {
      c0acc[v] = enc_c[0 * NB * H + (b0 + rb + v) * H + col];
      c1acc[v] = enc_c[1 * NB * H + (b0 + rb + v) * H + col];
    }
  }

  // layer1 weights resident in VGPRs: 32 fragments = 256 VGPRs per wave
  v16bf w1ih[4][4], w1hh[4][4];
  {
    const unsigned short* w1p = (const unsigned short*)(ws + WS_W1);
#pragma unroll
    for (int t = 0; t < 4; ++t)
#pragma unroll
      for (int kb = 0; kb < 4; ++kb) {
        int frag    = (wv + 8 * t) * 4 + kb;
        w1ih[t][kb] = loadB_pk(w1p, frag, lane);
        w1hh[t][kb] = loadB_pk(w1p + 65536, frag, lane);
      }
  }

  if (wv == 0) __builtin_amdgcn_s_wait_tensorcnt(0);  // TDM complete (wave0 issued it)
  __syncthreads();

  const unsigned short* sWih0 = sW0;
  const unsigned short* sWhh0 = sW0 + 65536;

#pragma unroll 1
  for (int step = 0; step < SEQ_LEN; ++step) {
    // layer0 t0: x=bufA, h=bufM1 -> bufM0
    lstm_cell<true>(lane, wv, bufA, bufM1, sWih0, sWhh0, nullptr, nullptr, sBS0, c0acc, bufM0, nullptr);
    __syncthreads();
    // layer0 t1: x=bufB, h=bufM0 -> bufM1
    lstm_cell<true>(lane, wv, bufB, bufM0, sWih0, sWhh0, nullptr, nullptr, sBS0, c0acc, bufM1, nullptr);
    __syncthreads();
    // layer1 t0: x=bufM0, h=bufH1 -> bufA
    lstm_cell<false>(lane, wv, bufM0, bufH1, nullptr, nullptr, w1ih, w1hh, sBS1, c1acc, bufA, nullptr);
    __syncthreads();
    // layer1 t1: x=bufM1, h=bufA -> bufB (+ bufH1 state)
    lstm_cell<false>(lane, wv, bufM1, bufA, nullptr, nullptr, w1ih, w1hh, sBS1, c1acc, bufB, bufH1);
    __syncthreads();
    // projection: sigmoid([bufA|bufB](16x256) @ WoutT) -> out[b, step, 0:12]
    if (wv == 0) {
      v8f acc = splat8(((lane & 15) < OUTF) ? sBout[lane & 15] : 0.f);
#pragma unroll
      for (int kb = 0; kb < 4; ++kb) {
        v16bf a0 = loadA_lds(bufA, kb, lane);
        acc = __builtin_amdgcn_wmma_f32_16x16x32_bf16(false, a0, false, loadB_pk(sWout, kb, lane),
                                                      (short)0, acc, false, false);
        v16bf a1 = loadA_lds(bufB, kb, lane);
        acc = __builtin_amdgcn_wmma_f32_16x16x32_bf16(false, a1, false, loadB_pk(sWout, 4 + kb, lane),
                                                      (short)0, acc, false, false);
      }
      int n = lane & 15;
      if (n < OUTF) {
        int rb = (lane < 16) ? 0 : 8;
#pragma unroll
        for (int v = 0; v < 8; ++v)
          out[(size_t)(b0 + rb + v) * (SEQ_LEN * OUTF) + step * OUTF + n] = sigm(acc[v]);
      }
    }
    __syncthreads();
  }
}

// ---------------- launch ----------------
extern "C" void kernel_launch(void* const* d_in, const int* in_sizes, int n_in,
                              void* d_out, int out_size, void* d_ws, size_t ws_size,
                              hipStream_t stream) {
  (void)in_sizes; (void)n_in; (void)out_size; (void)ws_size;
  const float* enc_h = (const float*)d_in[0];
  const float* enc_c = (const float*)d_in[1];
  const float* Wih0  = (const float*)d_in[2];
  const float* Whh0  = (const float*)d_in[3];
  const float* bih0  = (const float*)d_in[4];
  const float* bhh0  = (const float*)d_in[5];
  const float* Wih1  = (const float*)d_in[6];
  const float* Whh1  = (const float*)d_in[7];
  const float* bih1  = (const float*)d_in[8];
  const float* bhh1  = (const float*)d_in[9];
  const float* Wout  = (const float*)d_in[10];
  const float* bout  = (const float*)d_in[11];
  unsigned char* ws  = (unsigned char*)d_ws;

  pack_gates<<<1024, 256, 0, stream>>>(Wih0, Whh0, Wih1, Whh1,
                                       (unsigned short*)(ws + WS_W0));
  pack_wout<<<16, 256, 0, stream>>>(Wout, (unsigned short*)(ws + WS_WOUT));
  pack_bias<<<5, 256, 0, stream>>>(bih0, bhh0, bih1, bhh1, bout,
                                   (float*)(ws + WS_BS0));
  lstm_main<<<NB / 16, 256, 0, stream>>>(enc_h, enc_c, ws, (float*)d_out);
}
